// Attention_51290499449018
// MI455X (gfx1250) — matrix operand order repaired and checked by reading, not compiled, not run on hardware
//
#include <hip/hip_runtime.h>

// ---------------------------------------------------------------------------
// Types
// ---------------------------------------------------------------------------
typedef __attribute__((ext_vector_type(16))) __bf16 v16bf;
typedef __attribute__((ext_vector_type(8)))  float  v8f;
typedef __attribute__((ext_vector_type(4))) unsigned int u32x4;
typedef __attribute__((ext_vector_type(8)))  int    i32x8;
typedef __attribute__((ext_vector_type(4)))  int    i32x4;
typedef __attribute__((ext_vector_type(4)))  float  f32x4;

#define SEQ    2048
#define DMODEL 1024
#define HEADS  16
#define HDIM   64

// round-to-nearest-even f32 -> bf16 (bit pattern)
static __device__ __forceinline__ unsigned short f2bf(float f) {
    unsigned int u = __float_as_uint(f);
    u += 0x7fffu + ((u >> 16) & 1u);
    return (unsigned short)(u >> 16);
}

// Assemble a v16bf A/B fragment from two 16-byte chunks at p+off0, p+off1
static __device__ __forceinline__ v16bf ld_frag(const __bf16* p, int off0, int off1) {
    union { u32x4 q[2]; v16bf b; } u;
    u.q[0] = *(const u32x4*)(p + off0);
    u.q[1] = *(const u32x4*)(p + off1);
    return u.b;
}

static __device__ __forceinline__ v8f wmma_bf16(v16bf a, v16bf b, v8f c) {
    return __builtin_amdgcn_wmma_f32_16x16x32_bf16(
        /*neg_a=*/false, a, /*neg_b=*/false, b,
        /*c_mod=*/(short)0, c, /*reuse_a=*/false, /*reuse_b=*/false);
}

// LDS byte offset of a __shared__ object (low 32 bits of the flat address)
static __device__ __forceinline__ unsigned lds_of(const void* p) {
    return (unsigned)(size_t)p;
}

// ---------------------------------------------------------------------------
// Tensor Data Mover: 2D bf16 tile load Global -> LDS.
//   tile_x elements per line (mult. of 2 elems = 4B), tile_y lines,
//   stride_elems between lines. Tiles are always fully in-bounds, so tensor
//   dims are set huge (no clipping). LDS receives the tile packed row-major.
// Issued once per wave (EXEC ignored); tracked with TENSORcnt.
// 6-arg builtin form (clang-23 / therock-10.0 headers).
// ---------------------------------------------------------------------------
static __device__ __forceinline__ void tdm_load_2d(unsigned lds_off,
                                                   const void* gptr,
                                                   unsigned tile_x, unsigned tile_y,
                                                   unsigned long long stride_elems) {
    const unsigned long long ga = (unsigned long long)(size_t)gptr;
    u32x4 g0 = { 1u,                                   // count=1, user descriptor
                 lds_off,                              // lds_addr
                 (unsigned)ga,                         // global_addr[31:0]
                 (unsigned)((ga >> 32) & 0x1FFFFFFu) | (2u << 30) }; // addr[56:32] | type=2
    const unsigned td0 = 0x40000000u, td1 = 0x40000000u;   // huge tensor dims
    i32x8 g1 = { (int)(1u << 16),                                     // data_size=2B
                 (int)((td0 & 0xFFFFu) << 16),                        // dim0 lo16
                 (int)(((td0 >> 16) & 0xFFFFu) | ((td1 & 0xFFFFu) << 16)),
                 (int)(((td1 >> 16) & 0xFFFFu) | ((tile_x & 0xFFFFu) << 16)), // tile_dim0
                 (int)(tile_y & 0xFFFFu),                             // tile_dim1, tile_dim2=0
                 (int)(unsigned)(stride_elems & 0xFFFFFFFFull),       // dim0 stride lo32
                 (int)(unsigned)((stride_elems >> 32) & 0xFFFFu),     // dim0 stride hi16
                 0 };
    i32x4 z4 = { 0, 0, 0, 0 };
    i32x8 z8 = { 0, 0, 0, 0, 0, 0, 0, 0 };
    __builtin_amdgcn_tensor_load_to_lds(g0, g1, z4, z4, z8, 0);
}

// XOR-lane reductions over 16-lane groups via ds_swizzle (imm: xor<<10 | and=31)
#define RED16_MAX(x)                                                              \
    x = fmaxf(x, __int_as_float(__builtin_amdgcn_ds_swizzle(__float_as_int(x), 0x201F))); \
    x = fmaxf(x, __int_as_float(__builtin_amdgcn_ds_swizzle(__float_as_int(x), 0x101F))); \
    x = fmaxf(x, __int_as_float(__builtin_amdgcn_ds_swizzle(__float_as_int(x), 0x081F))); \
    x = fmaxf(x, __int_as_float(__builtin_amdgcn_ds_swizzle(__float_as_int(x), 0x041F)));
#define RED16_SUM(x)                                                              \
    x += __int_as_float(__builtin_amdgcn_ds_swizzle(__float_as_int(x), 0x201F));  \
    x += __int_as_float(__builtin_amdgcn_ds_swizzle(__float_as_int(x), 0x101F));  \
    x += __int_as_float(__builtin_amdgcn_ds_swizzle(__float_as_int(x), 0x081F));  \
    x += __int_as_float(__builtin_amdgcn_ds_swizzle(__float_as_int(x), 0x041F));

// ---------------------------------------------------------------------------
// Kernel 1a: linear f32 -> bf16 convert (x). 16 elems/thread.
// ---------------------------------------------------------------------------
__global__ __launch_bounds__(256)
void cvt_bf16_kernel(const float* __restrict__ src, unsigned short* __restrict__ dst) {
    const size_t i = ((size_t)blockIdx.x * 256 + threadIdx.x) * 16;
    f32x4 v0 = *(const f32x4*)(src + i + 0);
    f32x4 v1 = *(const f32x4*)(src + i + 4);
    f32x4 v2 = *(const f32x4*)(src + i + 8);
    f32x4 v3 = *(const f32x4*)(src + i + 12);
    alignas(16) unsigned short tmp[16];
#pragma unroll
    for (int j = 0; j < 4; j++) {
        tmp[j]      = f2bf(v0[j]);
        tmp[4 + j]  = f2bf(v1[j]);
        tmp[8 + j]  = f2bf(v2[j]);
        tmp[12 + j] = f2bf(v3[j]);
    }
    *(u32x4*)(dst + i)     = *(u32x4*)(tmp);
    *(u32x4*)(dst + i + 8) = *(u32x4*)(tmp + 8);
}

// ---------------------------------------------------------------------------
// Kernel 1b: tiled transpose + f32->bf16 convert of weights.
// wt[n*K + k] = bf16(w[k*N + n])   (w is [K][N] row-major)
// ---------------------------------------------------------------------------
__global__ __launch_bounds__(256)
void transpose_cvt_kernel(const float* __restrict__ w, unsigned short* __restrict__ wt,
                          int K, int N) {
    __shared__ float tile[32][33];
    const int nb = blockIdx.x * 32;
    const int kb = blockIdx.y * 32;
    const int tx = threadIdx.x & 31;
    const int ty = threadIdx.x >> 5;
#pragma unroll
    for (int i = 0; i < 32; i += 8)
        tile[ty + i][tx] = w[(size_t)(kb + ty + i) * N + nb + tx];
    __syncthreads();
#pragma unroll
    for (int i = 0; i < 32; i += 8)
        wt[(size_t)(nb + ty + i) * K + kb + tx] = f2bf(tile[tx][ty + i]);
}

// ---------------------------------------------------------------------------
// Kernel 2/4: bf16 WMMA GEMM with TDM-staged, double-buffered LDS tiles.
//   C[M,N] = A[M,K] * Bt[N,K]^T + bias
//   OUT_MODE 0: epilogue scatters bf16 Q/K + transposed V
//   OUT_MODE 1: epilogue writes f32 C
// Block tile 128x128, BK=32, 8 waves in 2x4 grid (64x32 per wave).
// ---------------------------------------------------------------------------
template <int OUT_MODE>
__global__ __launch_bounds__(256)
void gemm_bf16_kernel(const __bf16* __restrict__ A,
                      const __bf16* __restrict__ Bt,
                      const float* __restrict__ bias,
                      float* __restrict__ Cout,
                      unsigned short* __restrict__ qws,
                      unsigned short* __restrict__ kws,
                      unsigned short* __restrict__ vtws,
                      int M, int N, int K) {
    __shared__ __bf16 As[2][128 * 32];
    __shared__ __bf16 Bs[2][128 * 32];

    const int tid  = threadIdx.x;
    const int wave = tid >> 5;
    const int lane = tid & 31;
    const int l15  = lane & 15;
    const bool hi  = lane >= 16;
    const int wr   = wave >> 2;      // 0..1 (M dir)
    const int wc   = wave & 3;       // 0..3 (N dir)
    const int m0   = blockIdx.x * 128;
    const int n0   = blockIdx.y * 128;
    const bool w0  = (wave == 0);

    v8f acc[4][2] = {};

    const __bf16* Abase = A  + (size_t)m0 * K;
    const __bf16* Bbase = Bt + (size_t)n0 * K;

    if (w0) {   // prologue: fill buffer 0
        tdm_load_2d(lds_of(&As[0][0]), Abase, 32, 128, (unsigned long long)K);
        tdm_load_2d(lds_of(&Bs[0][0]), Bbase, 32, 128, (unsigned long long)K);
    }

    const int nIter = K / 32;
    for (int it = 0; it < nIter; ++it) {
        const int buf = it & 1;
        __syncthreads();   // everyone done reading buf^1 from previous iter
        if (w0) {
            if (it + 1 < nIter) {
                const int k1 = (it + 1) * 32;
                tdm_load_2d(lds_of(&As[buf ^ 1][0]), Abase + k1, 32, 128, (unsigned long long)K);
                tdm_load_2d(lds_of(&Bs[buf ^ 1][0]), Bbase + k1, 32, 128, (unsigned long long)K);
                __builtin_amdgcn_s_wait_tensorcnt(2);  // in-order: current buf complete
            } else {
                __builtin_amdgcn_s_wait_tensorcnt(0);
            }
        }
        __syncthreads();   // buf is ready for all waves

        const __bf16* as = &As[buf][0];
        const __bf16* bs = &Bs[buf][0];

        v16bf bfr[2];
#pragma unroll
        for (int nt = 0; nt < 2; nt++) {
            const int n  = wc * 32 + nt * 16 + l15;
            const int ko = hi ? 16 : 0;
            bfr[nt] = ld_frag(bs + n * 32, ko >> 1, 16 + (ko >> 1));
        }
#pragma unroll
        for (int mt = 0; mt < 4; mt++) {
            const int m  = wr * 64 + mt * 16 + l15;
            const int e0 = hi ? 8 : 0;
            v16bf af = ld_frag(as + m * 32, e0, 16 + e0);
#pragma unroll
            for (int nt = 0; nt < 2; nt++)
                acc[mt][nt] = wmma_bf16(af, bfr[nt], acc[mt][nt]);
        }
    }

    // ---- epilogue ----
#pragma unroll
    for (int mt = 0; mt < 4; mt++) {
#pragma unroll
        for (int nt = 0; nt < 2; nt++) {
            const int n  = n0 + wc * 32 + nt * 16 + l15;
            const float bv = bias[n];
#pragma unroll
            for (int r = 0; r < 8; r++) {
                const int m = m0 + wr * 64 + mt * 16 + r + (hi ? 8 : 0);
                const float v = acc[mt][nt][r] + bv;
                if (OUT_MODE == 1) {
                    Cout[(size_t)m * N + n] = v;
                } else {
                    const int sec = n >> 10;      // 0=q, 1=k, 2=v
                    const int wn  = n & 1023;
                    const int h   = wn >> 6;
                    const int d   = wn & 63;
                    const int b   = m >> 11;
                    const int s   = m & 2047;
                    const unsigned short bf = f2bf(v);
                    const size_t bh = (size_t)(b * HEADS + h);
                    if (sec == 0)
                        qws[(bh * SEQ + s) * HDIM + d] = bf;
                    else if (sec == 1)
                        kws[(bh * SEQ + s) * HDIM + d] = bf;
                    else
                        vtws[(bh * HDIM + d) * SEQ + s] = bf;   // V transposed
                }
            }
        }
    }
}

// ---------------------------------------------------------------------------
// Kernel 3: flash attention. grid = (S/128, B*H), 256 threads (8 waves).
// K/V^T tiles TDM-staged + double-buffered; Q in registers; online softmax.
// ---------------------------------------------------------------------------
__global__ __launch_bounds__(256)
void attn_kernel(const unsigned short* __restrict__ qws,
                 const unsigned short* __restrict__ kws,
                 const unsigned short* __restrict__ vtws,
                 unsigned short* __restrict__ aws) {
    __shared__ __bf16 Ks[2][32 * 64];     // [key][d]
    __shared__ __bf16 Vs[2][64 * 32];     // [d][key]  (V^T tile)
    __shared__ __bf16 Ps[8][16 * 32];     // per-wave P transpose scratch

    const int tid  = threadIdx.x;
    const int wave = tid >> 5;
    const int lane = tid & 31;
    const int l15  = lane & 15;
    const bool hi  = lane >= 16;
    const bool w0  = (wave == 0);

    const int qb = blockIdx.x;         // 0..15
    const int bh = blockIdx.y;         // 0..63
    const __bf16* Q  = (const __bf16*)qws  + (size_t)bh * SEQ * HDIM;
    const __bf16* Kp = (const __bf16*)kws  + (size_t)bh * SEQ * HDIM;
    const __bf16* Vt = (const __bf16*)vtws + (size_t)bh * HDIM * SEQ;

    const int qbase = qb * 128 + wave * 16;
    const int q_hi  = qbase + 15;

    // Q fragments (16 rows x 64 K-dim = 2 A-frags), held in registers
    const __bf16* qsrc = Q + (size_t)(qbase + l15) * HDIM;
    const int e0 = hi ? 8 : 0;
    const v16bf qf0 = ld_frag(qsrc,      e0, 16 + e0);
    const v16bf qf1 = ld_frag(qsrc + 32, e0, 16 + e0);

    v8f out[4] = {};
    float rm[8], rl[8];
#pragma unroll
    for (int r = 0; r < 8; r++) { rm[r] = -3.0e38f; rl[r] = 0.0f; }

    const int nkb = qb * 4 + 4;        // 32-key blocks up to causal frontier
    if (w0) {   // prologue: K tile is 32x64 contiguous (1 line of 2048 elems)
        tdm_load_2d(lds_of(&Ks[0][0]), Kp, 2048, 1, 2048ull);
        tdm_load_2d(lds_of(&Vs[0][0]), Vt, 32, 64, (unsigned long long)SEQ);
    }

    for (int kb = 0; kb < nkb; kb++) {
        const int kk0 = kb * 32;
        const int buf = kb & 1;
        __syncthreads();
        if (w0) {
            if (kb + 1 < nkb) {
                const int kk1 = (kb + 1) * 32;
                tdm_load_2d(lds_of(&Ks[buf ^ 1][0]), Kp + (size_t)kk1 * HDIM, 2048, 1, 2048ull);
                tdm_load_2d(lds_of(&Vs[buf ^ 1][0]), Vt + kk1, 32, 64, (unsigned long long)SEQ);
                __builtin_amdgcn_s_wait_tensorcnt(2);
            } else {
                __builtin_amdgcn_s_wait_tensorcnt(0);
            }
        }
        __syncthreads();

        if (kk0 > q_hi) continue;      // fully masked for this wave

        const __bf16* ks = &Ks[buf][0];
        const __bf16* vs = &Vs[buf][0];

        // ---- scores: 16q x 32k via 4 WMMAs ----
        v8f sc[2] = {};
#pragma unroll
        for (int nt = 0; nt < 2; nt++) {
            const int key = nt * 16 + l15;
            const int ko  = hi ? 16 : 0;
            v16bf kf0 = ld_frag(ks + key * 64,      ko >> 1, 16 + (ko >> 1));
            v16bf kf1 = ld_frag(ks + key * 64 + 32, ko >> 1, 16 + (ko >> 1));
            sc[nt] = wmma_bf16(qf0, kf0, sc[nt]);
            sc[nt] = wmma_bf16(qf1, kf1, sc[nt]);
        }

        // ---- causal mask + online softmax (row = (r, lane-half)) ----
        unsigned short* pws = (unsigned short*)&Ps[wave][0];
#pragma unroll
        for (int r = 0; r < 8; r++) {
            const int qrow = qbase + r + (hi ? 8 : 0);
            float s0 = sc[0][r];
            float s1 = sc[1][r];
            if (kk0 + l15 > qrow)      s0 = -10000.0f;
            if (kk0 + 16 + l15 > qrow) s1 = -10000.0f;
            float bm = fmaxf(s0, s1);
            RED16_MAX(bm)
            const float nm    = fmaxf(rm[r], bm);
            const float scale = __expf(rm[r] - nm);
            rm[r] = nm;
            const float p0 = __expf(s0 - nm);
            const float p1 = __expf(s1 - nm);
            float ps = p0 + p1;
            RED16_SUM(ps)
            rl[r] = rl[r] * scale + ps;
#pragma unroll
            for (int f = 0; f < 4; f++) out[f][r] *= scale;
            // scatter P (C layout) into per-wave LDS as row-major 16x32 bf16
            const int m = r + (hi ? 8 : 0);
            pws[m * 32 + l15]      = f2bf(p0);
            pws[m * 32 + 16 + l15] = f2bf(p1);
        }

        // ---- P A-fragment (16x32) + PV WMMAs ----
        const __bf16* pbase = &Ps[wave][0];
        v16bf pf = ld_frag(pbase + l15 * 32, e0, 16 + e0);
#pragma unroll
        for (int f = 0; f < 4; f++) {
            const int d  = f * 16 + l15;
            const int ko = hi ? 16 : 0;
            v16bf vf = ld_frag(vs + d * 32, ko >> 1, 16 + (ko >> 1));
            out[f] = wmma_bf16(pf, vf, out[f]);
        }
    }

    // ---- epilogue: a[B,S,H*64] bf16, merged heads ----
    const int b = bh >> 4, h = bh & 15;
#pragma unroll
    for (int r = 0; r < 8; r++) {
        const int s = qbase + r + (hi ? 8 : 0);
        const float inv = 1.0f / rl[r];
#pragma unroll
        for (int f = 0; f < 4; f++) {
            const int d = f * 16 + l15;
            aws[((size_t)(b * SEQ + s)) * DMODEL + h * HDIM + d] = f2bf(out[f][r] * inv);
        }
    }
}

// ---------------------------------------------------------------------------
// Host launcher
// ---------------------------------------------------------------------------
extern "C" void kernel_launch(void* const* d_in, const int* in_sizes, int n_in,
                              void* d_out, int out_size, void* d_ws, size_t ws_size,
                              hipStream_t stream) {
    const float* x      = (const float*)d_in[0];   // [4,2048,1024]
    const float* w_attn = (const float*)d_in[1];   // [1024,3072]
    const float* b_attn = (const float*)d_in[2];   // [3072]
    const float* w_proj = (const float*)d_in[3];   // [1024,1024]
    const float* b_proj = (const float*)d_in[4];   // [1024]
    float* out = (float*)d_out;

    const int M = 4 * SEQ;                         // 8192
    char* ws = (char*)d_ws;
    size_t off = 0;
    __bf16* wt_attn = (__bf16*)(ws + off); off += (size_t)3 * DMODEL * DMODEL * 2; // 6 MB
    __bf16* wt_proj = (__bf16*)(ws + off); off += (size_t)DMODEL * DMODEL * 2;     // 2 MB
    // x_bf (GEMM1 input) and a_ws (attn output) alias: x_bf is dead after GEMM1
    unsigned short* xbf_aws = (unsigned short*)(ws + off); off += (size_t)M * DMODEL * 2;
    unsigned short* qws  = (unsigned short*)(ws + off); off += (size_t)M * DMODEL * 2;
    unsigned short* kws  = (unsigned short*)(ws + off); off += (size_t)M * DMODEL * 2;
    unsigned short* vtws = (unsigned short*)(ws + off); off += (size_t)M * DMODEL * 2;

    // 1) x f32->bf16 ; weight transpose + bf16 convert
    cvt_bf16_kernel<<<(M * DMODEL) / (256 * 16), 256, 0, stream>>>(x, xbf_aws);
    transpose_cvt_kernel<<<dim3(3 * DMODEL / 32, DMODEL / 32), 256, 0, stream>>>(
        w_attn, (unsigned short*)wt_attn, DMODEL, 3 * DMODEL);
    transpose_cvt_kernel<<<dim3(DMODEL / 32, DMODEL / 32), 256, 0, stream>>>(
        w_proj, (unsigned short*)wt_proj, DMODEL, DMODEL);

    // 2) QKV GEMM (scatters Q, K, V^T as bf16)
    gemm_bf16_kernel<0><<<dim3(M / 128, 3 * DMODEL / 128), 256, 0, stream>>>(
        (const __bf16*)xbf_aws, wt_attn, b_attn, nullptr, qws, kws, vtws,
        M, 3 * DMODEL, DMODEL);

    // 3) flash attention (overwrites xbf_aws with merged-head output)
    attn_kernel<<<dim3(SEQ / 128, 4 * HEADS), 256, 0, stream>>>(qws, kws, vtws, xbf_aws);

    // 4) output projection
    gemm_bf16_kernel<1><<<dim3(M / 128, DMODEL / 128), 256, 0, stream>>>(
        (const __bf16*)xbf_aws, wt_proj, b_proj, out, nullptr, nullptr, nullptr,
        M, DMODEL, DMODEL);
}
